// GraphNetwork_5239860101622
// MI455X (gfx1250) — matrix operand order repaired and checked by reading, not compiled, not run on hardware
//
#include <hip/hip_runtime.h>
#include <stdint.h>

// ---------------- problem constants ----------------
#define NN 20000
#define EE 640000
#define DD 128          // node dim
#define RR 50           // rbf count
#define RP 64           // rbf padded to 64 halves
#define HH 256          // hidden
#define LL 4            // layers
#define K1 (2*DD + RP)  // 320 : padded message-MLP input dim
#define KIN (2*DD + RR) // 306 : true message-MLP input dim

typedef __attribute__((ext_vector_type(16))) _Float16 v16h;
typedef __attribute__((ext_vector_type(8)))  float    v8f;
typedef _Float16 f16_t;

union Frag { v16h h; uint32_t u[8]; };
union H2   { uint32_t u; f16_t h[2]; };

__device__ __forceinline__ float silu_f(float x) {
    return x / (1.0f + __expf(-x));
}

// load one 16x32 f16 A-fragment from an LDS row-major tile (row stride = stride halves)
__device__ __forceinline__ void load_a_frag(Frag& a, const f16_t* smem, int row,
                                            int stride, int k0, int akb) {
    const uint32_t* arow = (const uint32_t*)(smem + (size_t)row * stride);
    #pragma unroll
    for (int i = 0; i < 4; ++i) a.u[i]     = arow[(k0 + 2 * akb) / 2 + i];
    #pragma unroll
    for (int i = 0; i < 4; ++i) a.u[4 + i] = arow[(k0 + 2 * akb) / 2 + 4 + i];
}

// load one 32x16 f16 B-fragment from a transposed (out-major) weight [Nout, Kpad]
__device__ __forceinline__ void load_b_frag(Frag& b, const f16_t* wt, int n,
                                            int kpad, int ks) {
    const uint32_t* brow = (const uint32_t*)(wt + (size_t)n * kpad + ks);
    #pragma unroll
    for (int i = 0; i < 8; ++i) b.u[i] = brow[i];
}

// ---------------- prep kernels ----------------

__global__ void k_embed(const int* __restrict__ nf, const float* __restrict__ emb,
                        float* __restrict__ h32, f16_t* __restrict__ h16) {
    int i = blockIdx.x * blockDim.x + threadIdx.x;
    if (i >= NN * DD) return;
    int n = i >> 7, d = i & (DD - 1);
    float v = emb[(size_t)nf[n] * DD + d];
    h32[i] = v;
    h16[i] = (f16_t)v;
}

__global__ void k_rbf(const float* __restrict__ ea, f16_t* __restrict__ ef) {
    int e = blockIdx.x * blockDim.x + threadIdx.x;
    if (e >= EE) return;
    float x = ea[(size_t)e * 3 + 0];
    float y = ea[(size_t)e * 3 + 1];
    float z = ea[(size_t)e * 3 + 2];
    float dist = sqrtf(x * x + y * y + z * z);
    const float step   = 5.0f / 49.0f;   // linspace(0, 5, 50)
    const float inv2s2 = 50.0f;          // 1/(2*sigma^2), sigma = 5/50
    f16_t* row = ef + (size_t)e * RP;
    #pragma unroll 8
    for (int r = 0; r < RP; ++r) {
        float v = 0.0f;
        if (r < RR) { float dm = dist - step * (float)r; v = __expf(-dm * dm * inv2s2); }
        row[r] = (f16_t)v;
    }
}

// transpose+convert weight [K, Nout] f32 (row major) -> [Nout, Kpad] f16, zero-padded K
__global__ void k_wconv(const float* __restrict__ w, f16_t* __restrict__ wt,
                        int K, int Nout, int Kpad) {
    int i = blockIdx.x * blockDim.x + threadIdx.x;
    if (i >= Nout * Kpad) return;
    int n = i / Kpad, k = i % Kpad;
    float v = (k < K) ? w[(size_t)k * Nout + n] : 0.0f;
    wt[i] = (f16_t)v;
}

__global__ void k_zerof(float* __restrict__ p, int n) {
    int i = blockIdx.x * blockDim.x + threadIdx.x;
    if (i < n) p[i] = 0.0f;
}

// ---------------- message kernel (WMMA) ----------------
// block = 256 threads = 8 waves, owns 64 edges.
// Wave tiling: 2 M-tiles x 4 N-tiles per wave (GEMM1), 2x2 (GEMM2) -> B loads/WMMA = 0.5.
__global__ __launch_bounds__(256, 1)
void k_msg(const int* __restrict__ ei,       // [2,E]
           const f16_t* __restrict__ h16,    // [N,128]
           const f16_t* __restrict__ ef16,   // [E,64]
           const f16_t* __restrict__ w1t,    // [256,320] transposed, K-padded
           const float* __restrict__ b1,     // [256]
           const f16_t* __restrict__ w2t,    // [128,256]
           const float* __restrict__ b2,     // [128]
           float* __restrict__ agg,          // [N,128]
           int nblocks)
{
    __shared__ __align__(16) f16_t smem[64 * K1];   // 40KB; reused as hidden [64x256]
    const int tid  = threadIdx.x;
    const int lane = tid & 31;
    const int wv   = tid >> 5;           // 0..7
    const int e0   = blockIdx.x * 64;

    const int* __restrict__ src = ei;
    const int* __restrict__ dst = ei + EE;

    // ---- stage mi = concat(h[src], h[dst], ef) as dwords; prefetch next tile's rbf ----
    {
        uint32_t* s32 = (uint32_t*)smem;
        const uint32_t* h32p = (const uint32_t*)h16;
        const uint32_t* efp  = (const uint32_t*)ef16;
        for (int idx = tid; idx < 64 * (K1 / 2); idx += 256) {
            int e = idx / (K1 / 2);
            int j = idx % (K1 / 2);      // dword col 0..159
            int eg = e0 + e;
            uint32_t v;
            if (j < 64)       v = h32p[(size_t)src[eg] * 64 + j];
            else if (j < 128) v = h32p[(size_t)dst[eg] * 64 + (j - 64)];
            else              v = efp[(size_t)eg * 32 + (j - 128)];
            s32[idx] = v;
        }
        // HBM stream: prefetch next tile's RBF rows (global_prefetch)
        if (blockIdx.x + 1 < (unsigned)nblocks && tid < 64) {
            __builtin_prefetch(ef16 + (size_t)(e0 + 64 + tid) * RP, 0, 1);
        }
    }
    __syncthreads();

    const int mgrp = wv & 1;                  // row group: rows mgrp*32 .. +31 (2 M-tiles)
    const int ngrp = wv >> 1;                 // col group: cols ngrp*64 .. +63 (4 N-tiles)
    const int lrow = lane & 15;
    const int akb  = (lane >> 4) * 8;         // A k sub-offset
    const int am0  = mgrp * 32 + lrow;        // A row, m-tile 0
    const int am1  = mgrp * 32 + 16 + lrow;   // A row, m-tile 1

    // ---- GEMM1: [64x320] @ W1^T -> [64x256], 10 k-steps ----
    v8f acc[2][4];
    #pragma unroll
    for (int mi_ = 0; mi_ < 2; ++mi_)
        #pragma unroll
        for (int ji = 0; ji < 4; ++ji) { v8f z = {}; acc[mi_][ji] = z; }

    for (int kt = 0; kt < K1 / 32; ++kt) {
        const int k0 = kt * 32;
        const int ks = k0 + (lane >> 4) * 16;
        Frag a[2], b[4];
        load_a_frag(a[0], smem, am0, K1, k0, akb);
        load_a_frag(a[1], smem, am1, K1, k0, akb);
        #pragma unroll
        for (int ji = 0; ji < 4; ++ji)
            load_b_frag(b[ji], w1t, ngrp * 64 + ji * 16 + lrow, K1, ks);
        #pragma unroll
        for (int mi_ = 0; mi_ < 2; ++mi_)
            #pragma unroll
            for (int ji = 0; ji < 4; ++ji)
                acc[mi_][ji] = __builtin_amdgcn_wmma_f32_16x16x32_f16(
                    false, a[mi_].h, false, b[ji].h, (short)0, acc[mi_][ji], false, false);
    }
    __syncthreads();   // all LDS reads of mi complete

    // ---- bias + silu -> hidden in LDS (overlaid) ----
    {
        #pragma unroll
        for (int mi_ = 0; mi_ < 2; ++mi_) {
            const int rbase = mgrp * 32 + mi_ * 16 + (lane >> 4) * 8;
            #pragma unroll
            for (int ji = 0; ji < 4; ++ji) {
                const int n = ngrp * 64 + ji * 16 + lrow;
                const float bias = b1[n];
                #pragma unroll
                for (int r = 0; r < 8; ++r) {
                    float v = silu_f(acc[mi_][ji][r] + bias);
                    smem[(size_t)(rbase + r) * HH + n] = (f16_t)v;
                }
            }
        }
    }
    __syncthreads();

    // ---- GEMM2: [64x256] @ W2^T -> [64x128], 8 k-steps, 2x2 tiles/wave ----
    v8f acc2[2][2];
    #pragma unroll
    for (int mi_ = 0; mi_ < 2; ++mi_)
        #pragma unroll
        for (int ji = 0; ji < 2; ++ji) { v8f z = {}; acc2[mi_][ji] = z; }

    for (int kt = 0; kt < HH / 32; ++kt) {
        const int k0 = kt * 32;
        const int ks = k0 + (lane >> 4) * 16;
        Frag a[2], b[2];
        load_a_frag(a[0], smem, am0, HH, k0, akb);
        load_a_frag(a[1], smem, am1, HH, k0, akb);
        #pragma unroll
        for (int ji = 0; ji < 2; ++ji)
            load_b_frag(b[ji], w2t, ngrp * 32 + ji * 16 + lrow, HH, ks);
        #pragma unroll
        for (int mi_ = 0; mi_ < 2; ++mi_)
            #pragma unroll
            for (int ji = 0; ji < 2; ++ji)
                acc2[mi_][ji] = __builtin_amdgcn_wmma_f32_16x16x32_f16(
                    false, a[mi_].h, false, b[ji].h, (short)0, acc2[mi_][ji], false, false);
    }

    // ---- bias + atomic scatter into agg[dst] (L2 resident) ----
    {
        #pragma unroll
        for (int mi_ = 0; mi_ < 2; ++mi_) {
            const int rbase = mgrp * 32 + mi_ * 16 + (lane >> 4) * 8;
            #pragma unroll
            for (int ji = 0; ji < 2; ++ji) {
                const int n = ngrp * 32 + ji * 16 + lrow;
                const float bias = b2[n];
                #pragma unroll
                for (int r = 0; r < 8; ++r) {
                    int eg = e0 + rbase + r;
                    int dn = dst[eg];
                    atomicAdd(&agg[(size_t)dn * DD + n], acc2[mi_][ji][r] + bias);
                }
            }
        }
    }
}

// ---------------- node-update kernel (WMMA) ----------------
// block = 256 threads = 8 waves, owns 64 nodes.
__global__ __launch_bounds__(256, 1)
void k_upd(f16_t* __restrict__ h16,          // [N,128] read + write
           const float* __restrict__ agg,    // [N,128]
           const f16_t* __restrict__ u1t,    // [256,256]
           const float* __restrict__ b1,     // [256]
           const f16_t* __restrict__ u2t,    // [128,256]
           const float* __restrict__ b2,     // [128]
           float* __restrict__ h32)          // [N,128] read + write
{
    __shared__ __align__(16) f16_t smem[64 * HH];   // 32KB; ui then hidden
    const int tid  = threadIdx.x;
    const int lane = tid & 31;
    const int wv   = tid >> 5;
    const int n0   = blockIdx.x * 64;

    // ---- stage ui = concat(h, agg) ----
    {
        uint32_t* s32 = (uint32_t*)smem;
        const uint32_t* h32p = (const uint32_t*)h16;
        for (int idx = tid; idx < 64 * (HH / 2); idx += 256) {
            int row = idx / (HH / 2);
            int j   = idx % (HH / 2);     // dword col 0..127
            int node = n0 + row;
            uint32_t v = 0u;
            if (node < NN) {
                if (j < 64) {
                    v = h32p[(size_t)node * 64 + j];
                } else {
                    const float* ap = agg + (size_t)node * DD + (size_t)(j - 64) * 2;
                    H2 p; p.h[0] = (f16_t)ap[0]; p.h[1] = (f16_t)ap[1];
                    v = p.u;
                }
            }
            s32[idx] = v;
        }
    }
    __syncthreads();

    const int mgrp = wv & 1;
    const int ngrp = wv >> 1;
    const int lrow = lane & 15;
    const int akb  = (lane >> 4) * 8;
    const int am0  = mgrp * 32 + lrow;
    const int am1  = mgrp * 32 + 16 + lrow;

    // ---- GEMM1: [64x256] @ U1^T -> [64x256], 8 k-steps ----
    v8f acc[2][4];
    #pragma unroll
    for (int mi_ = 0; mi_ < 2; ++mi_)
        #pragma unroll
        for (int ji = 0; ji < 4; ++ji) { v8f z = {}; acc[mi_][ji] = z; }

    for (int kt = 0; kt < HH / 32; ++kt) {
        const int k0 = kt * 32;
        const int ks = k0 + (lane >> 4) * 16;
        Frag a[2], b[4];
        load_a_frag(a[0], smem, am0, HH, k0, akb);
        load_a_frag(a[1], smem, am1, HH, k0, akb);
        #pragma unroll
        for (int ji = 0; ji < 4; ++ji)
            load_b_frag(b[ji], u1t, ngrp * 64 + ji * 16 + lrow, HH, ks);
        #pragma unroll
        for (int mi_ = 0; mi_ < 2; ++mi_)
            #pragma unroll
            for (int ji = 0; ji < 4; ++ji)
                acc[mi_][ji] = __builtin_amdgcn_wmma_f32_16x16x32_f16(
                    false, a[mi_].h, false, b[ji].h, (short)0, acc[mi_][ji], false, false);
    }
    __syncthreads();

    {
        #pragma unroll
        for (int mi_ = 0; mi_ < 2; ++mi_) {
            const int rbase = mgrp * 32 + mi_ * 16 + (lane >> 4) * 8;
            #pragma unroll
            for (int ji = 0; ji < 4; ++ji) {
                const int n = ngrp * 64 + ji * 16 + lrow;
                const float bias = b1[n];
                #pragma unroll
                for (int r = 0; r < 8; ++r) {
                    float v = silu_f(acc[mi_][ji][r] + bias);
                    smem[(size_t)(rbase + r) * HH + n] = (f16_t)v;
                }
            }
        }
    }
    __syncthreads();

    // ---- GEMM2: [64x256] @ U2^T -> [64x128], 8 k-steps, 2x2 tiles/wave ----
    v8f acc2[2][2];
    #pragma unroll
    for (int mi_ = 0; mi_ < 2; ++mi_)
        #pragma unroll
        for (int ji = 0; ji < 2; ++ji) { v8f z = {}; acc2[mi_][ji] = z; }

    for (int kt = 0; kt < HH / 32; ++kt) {
        const int k0 = kt * 32;
        const int ks = k0 + (lane >> 4) * 16;
        Frag a[2], b[2];
        load_a_frag(a[0], smem, am0, HH, k0, akb);
        load_a_frag(a[1], smem, am1, HH, k0, akb);
        #pragma unroll
        for (int ji = 0; ji < 2; ++ji)
            load_b_frag(b[ji], u2t, ngrp * 32 + ji * 16 + lrow, HH, ks);
        #pragma unroll
        for (int mi_ = 0; mi_ < 2; ++mi_)
            #pragma unroll
            for (int ji = 0; ji < 2; ++ji)
                acc2[mi_][ji] = __builtin_amdgcn_wmma_f32_16x16x32_f16(
                    false, a[mi_].h, false, b[ji].h, (short)0, acc2[mi_][ji], false, false);
    }

    // ---- residual: h += out ----
    {
        #pragma unroll
        for (int mi_ = 0; mi_ < 2; ++mi_) {
            const int rbase = mgrp * 32 + mi_ * 16 + (lane >> 4) * 8;
            #pragma unroll
            for (int ji = 0; ji < 2; ++ji) {
                const int n = ngrp * 32 + ji * 16 + lrow;
                const float bias = b2[n];
                #pragma unroll
                for (int r = 0; r < 8; ++r) {
                    int node = n0 + rbase + r;
                    if (node < NN) {
                        size_t off = (size_t)node * DD + n;
                        float hv = h32[off] + acc2[mi_][ji][r] + bias;
                        h32[off] = hv;
                        h16[off] = (f16_t)hv;
                    }
                }
            }
        }
    }
}

// ---------------- output head + reduction ----------------
__global__ __launch_bounds__(256)
void k_out(const float* __restrict__ h32, const float* __restrict__ w1,
           const float* __restrict__ b1, const float* __restrict__ w2,
           const float* __restrict__ b2, float* __restrict__ out) {
    int n = blockIdx.x * blockDim.x + threadIdx.x;
    float part = 0.0f;
    if (n < NN) {
        float s[64];
        #pragma unroll
        for (int d = 0; d < 64; ++d) s[d] = b1[d];
        const float* hr = h32 + (size_t)n * DD;
        for (int k = 0; k < DD; ++k) {
            float hv = hr[k];
            const float* wr = w1 + (size_t)k * 64;
            #pragma unroll
            for (int d = 0; d < 64; ++d) s[d] += hv * wr[d];
        }
        float acc = b2[0];
        #pragma unroll
        for (int d = 0; d < 64; ++d) acc += silu_f(s[d]) * w2[d];
        part = acc;
    }
    __shared__ float red[256];
    red[threadIdx.x] = part;
    __syncthreads();
    for (int s = 128; s > 0; s >>= 1) {
        if (threadIdx.x < s) red[threadIdx.x] += red[threadIdx.x + s];
        __syncthreads();
    }
    if (threadIdx.x == 0) atomicAdd(out, red[0]);
}

// ---------------- host launcher ----------------
extern "C" void kernel_launch(void* const* d_in, const int* in_sizes, int n_in,
                              void* d_out, int out_size, void* d_ws, size_t ws_size,
                              hipStream_t stream) {
    (void)in_sizes; (void)n_in; (void)out_size; (void)ws_size;
    const int*   nf    = (const int*)  d_in[0];
    const int*   ei    = (const int*)  d_in[1];
    const float* ea    = (const float*)d_in[2];
    const float* emb   = (const float*)d_in[3];
    const float* msgW1 = (const float*)d_in[4];
    const float* msgb1 = (const float*)d_in[5];
    const float* msgW2 = (const float*)d_in[6];
    const float* msgb2 = (const float*)d_in[7];
    const float* updW1 = (const float*)d_in[8];
    const float* updb1 = (const float*)d_in[9];
    const float* updW2 = (const float*)d_in[10];
    const float* updb2 = (const float*)d_in[11];
    const float* outW1 = (const float*)d_in[12];
    const float* outb1 = (const float*)d_in[13];
    const float* outW2 = (const float*)d_in[14];
    const float* outb2 = (const float*)d_in[15];
    float* out = (float*)d_out;

    // workspace layout (~104 MB, all chunks 16B aligned)
    char* p = (char*)d_ws;
    float* h32  = (float*)p;  p += (size_t)NN * DD * 4;
    float* agg  = (float*)p;  p += (size_t)NN * DD * 4;
    f16_t* h16  = (f16_t*)p;  p += (size_t)NN * DD * 2;
    f16_t* ef16 = (f16_t*)p;  p += (size_t)EE * RP * 2;
    f16_t* w1t  = (f16_t*)p;  p += (size_t)LL * HH * K1 * 2;
    f16_t* w2t  = (f16_t*)p;  p += (size_t)LL * DD * HH * 2;
    f16_t* u1t  = (f16_t*)p;  p += (size_t)LL * HH * HH * 2;
    f16_t* u2t  = (f16_t*)p;  p += (size_t)LL * DD * HH * 2;

    // prep
    k_embed<<<(NN * DD + 255) / 256, 256, 0, stream>>>(nf, emb, h32, h16);
    k_rbf<<<(EE + 255) / 256, 256, 0, stream>>>(ea, ef16);
    for (int l = 0; l < LL; ++l) {
        k_wconv<<<(HH * K1 + 255) / 256, 256, 0, stream>>>(
            msgW1 + (size_t)l * KIN * HH, w1t + (size_t)l * HH * K1, KIN, HH, K1);
        k_wconv<<<(DD * HH + 255) / 256, 256, 0, stream>>>(
            msgW2 + (size_t)l * HH * DD, w2t + (size_t)l * DD * HH, HH, DD, HH);
        k_wconv<<<(HH * HH + 255) / 256, 256, 0, stream>>>(
            updW1 + (size_t)l * HH * HH, u1t + (size_t)l * HH * HH, HH, HH, HH);
        k_wconv<<<(DD * HH + 255) / 256, 256, 0, stream>>>(
            updW2 + (size_t)l * HH * DD, u2t + (size_t)l * DD * HH, HH, DD, HH);
    }

    // layers
    const int msg_blocks = EE / 64;
    for (int l = 0; l < LL; ++l) {
        k_zerof<<<(NN * DD + 255) / 256, 256, 0, stream>>>(agg, NN * DD);
        k_msg<<<msg_blocks, 256, 0, stream>>>(
            ei, h16, ef16,
            w1t + (size_t)l * HH * K1, msgb1 + (size_t)l * HH,
            w2t + (size_t)l * DD * HH, msgb2 + (size_t)l * DD,
            agg, msg_blocks);
        k_upd<<<(NN + 63) / 64, 256, 0, stream>>>(
            h16, agg,
            u1t + (size_t)l * HH * HH, updb1 + (size_t)l * HH,
            u2t + (size_t)l * DD * HH, updb2 + (size_t)l * DD,
            h32);
    }

    // output head
    k_zerof<<<1, 32, 0, stream>>>(out, 1);
    k_out<<<(NN + 255) / 256, 256, 0, stream>>>(h32, outW1, outb1, outW2, outb2, out);
}
